// TransformerBlock_89172110999924
// MI455X (gfx1250) — matrix operand order repaired and checked by reading, not compiled, not run on hardware
//
#include <hip/hip_runtime.h>
#include <hip/hip_bf16.h>

// ---------------------------------------------------------------------------
// CDNA5 (gfx1250) transformer block, bf16 WMMA (f32 accumulate), wave32.
// Register-blocked 32x32 wave tiles: 4 WMMAs per 4 fragment loads.
// ---------------------------------------------------------------------------

typedef __bf16 bf16_t;
typedef __attribute__((ext_vector_type(16))) __bf16 bf16x16;
typedef __attribute__((ext_vector_type(8)))  __bf16 bf16x8;
typedef __attribute__((ext_vector_type(8)))  float  f32x8;

#define B_ 8
#define S_ 2048
#define E_ 1024
#define F_ 2048
#define M_ (B_ * S_)

// ---- fp32 -> bf16 (round-to-nearest-even) ---------------------------------
__device__ __forceinline__ bf16_t f2bf(float f) {
  unsigned u = __builtin_bit_cast(unsigned, f);
  u += 0x7FFFu + ((u >> 16) & 1u);
  unsigned short h = (unsigned short)(u >> 16);
  return __builtin_bit_cast(bf16_t, h);
}

// ---- WMMA fragment loaders (contiguous k order, same for A and B) -----
// A: 16x32 bf16, row-major source, lda in elements.
// lane<16 : row=lane,    K = k0+0..15 contiguous
// lane>=16: row=lane-16, K = k0+16..31 contiguous
__device__ __forceinline__ bf16x16 load_a_frag(const bf16_t* A, int lda,
                                               int m0, int k0, int lane) {
  const int r = m0 + (lane & 15);
  const int c = k0 + ((lane >> 4) << 4);
  const bf16_t* p = A + (size_t)r * lda + c;
  return *(const bf16x16*)(p);
}

// B: 32x16 bf16 (KxN). Source stored [N][K] row-major (ldb elements per N row).
// lane<16 : col=lane,    K = k0+0..15 contiguous
// lane>=16: col=lane-16, K = k0+16..31 contiguous
__device__ __forceinline__ bf16x16 load_b_frag(const bf16_t* Bm, int ldb,
                                               int n0, int k0, int lane) {
  const int n = n0 + (lane & 15);
  const int c = k0 + ((lane >> 4) << 4);
  return *(const bf16x16*)(Bm + (size_t)n * ldb + c);
}

__device__ __forceinline__ f32x8 wmma_bf16(bf16x16 a, bf16x16 b, f32x8 c) {
  return __builtin_amdgcn_wmma_f32_16x16x32_bf16(false, a, false, b,
                                                 (short)0, c, false, false);
}

// ---- fp32 -> bf16 bulk convert (weights) ----------------------------------
__global__ void cvt_kernel(const float* __restrict__ src,
                           bf16_t* __restrict__ dst, int n) {
  int i = blockIdx.x * blockDim.x + threadIdx.x;
  if (i < n) dst[i] = f2bf(src[i]);
}

// ---- LayerNorm over E=1024, one block (8 waves) per row, bf16 out ---------
__global__ __launch_bounds__(256) void ln_kernel(const float* __restrict__ x,
                                                 const float* __restrict__ w,
                                                 const float* __restrict__ b,
                                                 bf16_t* __restrict__ out) {
  const int row = blockIdx.x;
  const float* xr = x + (size_t)row * E_;
  __shared__ float red[2][8];
  float s = 0.f, ss = 0.f;
  for (int i = threadIdx.x; i < E_; i += 256) {
    float v = xr[i];
    s += v; ss += v * v;
  }
#pragma unroll
  for (int o = 16; o; o >>= 1) {
    s += __shfl_xor(s, o, 32);
    ss += __shfl_xor(ss, o, 32);
  }
  const int wid = threadIdx.x >> 5;
  if ((threadIdx.x & 31) == 0) { red[0][wid] = s; red[1][wid] = ss; }
  __syncthreads();
  float st = 0.f, sst = 0.f;
#pragma unroll
  for (int i = 0; i < 8; ++i) { st += red[0][i]; sst += red[1][i]; }
  const float mu = st * (1.0f / E_);
  const float var = sst * (1.0f / E_) - mu * mu;
  const float rinv = rsqrtf(var + 1e-5f);
  for (int i = threadIdx.x; i < E_; i += 256) {
    float v = (xr[i] - mu) * rinv * w[i] + b[i];
    out[(size_t)row * E_ + i] = f2bf(v);
  }
}

// ---- GEMM epilogue for one 16x16 C tile -----------------------------------
// mode 0: fp32 out (+bias +residual)         -> outF
// mode 1: bf16 row-major out (+bias, ?relu)  -> outB
// mode 2: bf16 transposed per-batch out (V)  -> outB[b][n][s], s-major
__device__ __forceinline__ void store_tile(
    f32x8 acc, int m0, int n0, int lane, const float* bias,
    const float* residual, float* outF, bf16_t* outB, int N, int mode,
    int relu) {
  const int cn = n0 + (lane & 15);
  const int rb = (lane >> 4) * 8;
  const float bvv = bias ? bias[cn] : 0.f;
#pragma unroll
  for (int v = 0; v < 8; ++v) {
    const int m = m0 + rb + v;
    float val = acc[v] + bvv;
    if (relu) val = fmaxf(val, 0.f);
    if (mode == 0) {
      const size_t idx = (size_t)m * N + cn;
      outF[idx] = val + (residual ? residual[idx] : 0.f);
    } else if (mode == 1) {
      outB[(size_t)m * N + cn] = f2bf(val);
    } else {  // transposed V store: Vt[b][n][s]
      const int bb = m / S_, ss2 = m - bb * S_;
      outB[((size_t)bb * N + cn) * (size_t)S_ + ss2] = f2bf(val);
    }
  }
}

// ---- WMMA GEMM, 32x32 tile per wave:  C = A[M,K] * Bw[N,K]^T --------------
__global__ __launch_bounds__(256) void gemm_kernel(
    const bf16_t* __restrict__ A, const bf16_t* __restrict__ Bw,
    const float* __restrict__ bias, const float* __restrict__ residual,
    float* __restrict__ outF, bf16_t* __restrict__ outB,
    int M, int N, int K, int mode, int relu) {
  const int wid = (blockIdx.x * 256 + threadIdx.x) >> 5;  // global wave id
  const int lane = threadIdx.x & 31;
  const int tn2 = N >> 5;
  const int mt = wid / tn2;
  const int nt = wid - mt * tn2;
  if (mt >= (M >> 5)) return;  // wave-uniform guard (EXEC stays all-1 for WMMA)
  const int m0 = mt << 5, n0 = nt << 5;

  f32x8 c00 = {}, c01 = {}, c10 = {}, c11 = {};
  for (int k = 0; k < K; k += 32) {
    bf16x16 a0 = load_a_frag(A, K, m0, k, lane);
    bf16x16 a1 = load_a_frag(A, K, m0 + 16, k, lane);
    bf16x16 b0 = load_b_frag(Bw, K, n0, k, lane);
    bf16x16 b1 = load_b_frag(Bw, K, n0 + 16, k, lane);
    c00 = wmma_bf16(a0, b0, c00);
    c01 = wmma_bf16(a0, b1, c01);
    c10 = wmma_bf16(a1, b0, c10);
    c11 = wmma_bf16(a1, b1, c11);
  }
  store_tile(c00, m0, n0, lane, bias, residual, outF, outB, N, mode, relu);
  store_tile(c01, m0, n0 + 16, lane, bias, residual, outF, outB, N, mode, relu);
  store_tile(c10, m0 + 16, n0, lane, bias, residual, outF, outB, N, mode, relu);
  store_tile(c11, m0 + 16, n0 + 16, lane, bias, residual, outF, outB, N, mode,
             relu);
}

// ---- fused attention for one (batch, 16 query rows) block -----------------
// scores (16 x S) live entirely in LDS (bf16, 64KB); never touch HBM.
__global__ __launch_bounds__(256) void attn_kernel(
    const bf16_t* __restrict__ Q, const bf16_t* __restrict__ Kb,
    const bf16_t* __restrict__ Vt, const unsigned char* __restrict__ mask,
    const float* __restrict__ x, float* __restrict__ x2) {
  __shared__ bf16_t sc[16 * S_];  // 64 KB

  const int b = blockIdx.x / (S_ / 16);
  const int q0 = (blockIdx.x % (S_ / 16)) * 16;
  const int wid = threadIdx.x >> 5;
  const int lane = threadIdx.x & 31;

  // ---- phase 1: scores (16 x S) = Q_tile (16xE) * K^T; 2 K-tiles/A-frag --
  const bf16_t* Qrow = Q + ((size_t)(b * S_ + q0)) * E_;
  const bf16_t* Krow = Kb + (size_t)b * S_ * E_;  // [k][d] == B layout [N][K]
  for (int nt = wid * 2; nt < S_ / 16; nt += 16) {
    f32x8 acc0 = {}, acc1 = {};
    for (int k = 0; k < E_; k += 32) {
      bf16x16 a = load_a_frag(Qrow, E_, 0, k, lane);
      bf16x16 b0 = load_b_frag(Krow, E_, nt * 16, k, lane);
      bf16x16 b1 = load_b_frag(Krow, E_, nt * 16 + 16, k, lane);
      acc0 = wmma_bf16(a, b0, acc0);
      acc1 = wmma_bf16(a, b1, acc1);
    }
    const int cn = nt * 16 + (lane & 15);
    const int rb = (lane >> 4) * 8;
#pragma unroll
    for (int v = 0; v < 8; ++v) {
      sc[(rb + v) * S_ + cn] = f2bf(acc0[v]);
      sc[(rb + v) * S_ + cn + 16] = f2bf(acc1[v]);
    }
  }
  __syncthreads();

  // ---- phase 2: masked softmax over each of the 16 rows (16 lanes/row) ---
  {
    const int row = threadIdx.x >> 4;
    const int t16 = threadIdx.x & 15;
    const unsigned char* mrow = mask + (size_t)b * S_;
    const float scale = 0.03125f;  // 1/sqrt(1024)
    float mx = -__builtin_inff();
    for (int c = t16; c < S_; c += 16) {
      float v = (float)sc[row * S_ + c] * scale;
      if (mrow[c]) v = -__builtin_inff();
      sc[row * S_ + c] = f2bf(v);
      mx = fmaxf(mx, v);
    }
#pragma unroll
    for (int o = 8; o; o >>= 1) mx = fmaxf(mx, __shfl_xor(mx, o, 16));
    float sum = 0.f;
    for (int c = t16; c < S_; c += 16) {
      float e = __expf((float)sc[row * S_ + c] - mx);
      sc[row * S_ + c] = f2bf(e);
      sum += e;
    }
#pragma unroll
    for (int o = 8; o; o >>= 1) sum += __shfl_xor(sum, o, 16);
    const float rs = 1.0f / sum;
    for (int c = t16; c < S_; c += 16)
      sc[row * S_ + c] = f2bf((float)sc[row * S_ + c] * rs);
  }
  __syncthreads();

  // ---- phase 3: out (16 x E) = probs (16 x S) * V; 2 d-tiles/A-frag ------
  const bf16_t* Vb = Vt + (size_t)b * E_ * S_;  // [d][s] == B layout [N][K]
  for (int dt = wid * 2; dt < E_ / 16; dt += 16) {
    f32x8 acc0 = {}, acc1 = {};
    for (int k = 0; k < S_; k += 32) {
      bf16x16 a = load_a_frag(sc, S_, 0, k, lane);  // probs from LDS
      bf16x16 b0 = load_b_frag(Vb, S_, dt * 16, k, lane);
      bf16x16 b1 = load_b_frag(Vb, S_, dt * 16 + 16, k, lane);
      acc0 = wmma_bf16(a, b0, acc0);
      acc1 = wmma_bf16(a, b1, acc1);
    }
    const int cn = dt * 16 + (lane & 15);
    const int rb = (lane >> 4) * 8;
#pragma unroll
    for (int v = 0; v < 8; ++v) {
      const size_t idx = ((size_t)(b * S_ + q0 + rb + v)) * E_ + cn;
      x2[idx] = x[idx] + acc0[v];
      x2[idx + 16] = x[idx + 16] + acc1[v];
    }
  }
}

// ---------------------------------------------------------------------------
extern "C" void kernel_launch(void* const* d_in, const int* in_sizes, int n_in,
                              void* d_out, int out_size, void* d_ws,
                              size_t ws_size, hipStream_t stream) {
  (void)in_sizes; (void)n_in; (void)out_size; (void)ws_size;
  const float* x = (const float*)d_in[0];
  const unsigned char* mask = (const unsigned char*)d_in[1];
  const float* ln1w = (const float*)d_in[2];
  const float* ln1b = (const float*)d_in[3];
  const float* ln2w = (const float*)d_in[4];
  const float* ln2b = (const float*)d_in[5];
  const float* Wq = (const float*)d_in[6];
  const float* bq = (const float*)d_in[7];
  const float* Wk = (const float*)d_in[8];
  const float* bk = (const float*)d_in[9];
  const float* Wv = (const float*)d_in[10];
  const float* bv = (const float*)d_in[11];
  const float* W1 = (const float*)d_in[12];
  const float* b1 = (const float*)d_in[13];
  const float* W2 = (const float*)d_in[14];
  const float* b2 = (const float*)d_in[15];
  float* out = (float*)d_out;

  char* ws = (char*)d_ws;
  const size_t MB = 1ull << 20;
  bf16_t* normed = (bf16_t*)(ws + 0 * MB);    // 32MB (reused for LN2 output)
  bf16_t* Wq_b   = (bf16_t*)(ws + 32 * MB);   // 2MB
  bf16_t* Wk_b   = (bf16_t*)(ws + 34 * MB);   // 2MB
  bf16_t* Wv_b   = (bf16_t*)(ws + 36 * MB);   // 2MB
  bf16_t* W1_b   = (bf16_t*)(ws + 38 * MB);   // 4MB
  bf16_t* W2_b   = (bf16_t*)(ws + 42 * MB);   // 4MB
  bf16_t* Qb     = (bf16_t*)(ws + 46 * MB);   // 32MB
  bf16_t* Kbuf   = (bf16_t*)(ws + 78 * MB);   // 32MB
  bf16_t* Vt     = (bf16_t*)(ws + 110 * MB);  // 32MB  [b][d][s]
  float*  x2     = (float*)(ws + 142 * MB);   // 64MB
  bf16_t* h      = (bf16_t*)(ws + 46 * MB);   // 64MB, reuses Q+K after attn

  // 1) weights -> bf16 (layouts already [N][K] for B-fragments)
  cvt_kernel<<<(E_ * E_ + 255) / 256, 256, 0, stream>>>(Wq, Wq_b, E_ * E_);
  cvt_kernel<<<(E_ * E_ + 255) / 256, 256, 0, stream>>>(Wk, Wk_b, E_ * E_);
  cvt_kernel<<<(E_ * E_ + 255) / 256, 256, 0, stream>>>(Wv, Wv_b, E_ * E_);
  cvt_kernel<<<(F_ * E_ + 255) / 256, 256, 0, stream>>>(W1, W1_b, F_ * E_);
  cvt_kernel<<<(E_ * F_ + 255) / 256, 256, 0, stream>>>(W2, W2_b, E_ * F_);

  // 2) LN1
  ln_kernel<<<M_, 256, 0, stream>>>(x, ln1w, ln1b, normed);

  // 3) Q/K/V projections (WMMA, 32x32 wave tiles). V written transposed.
  {
    const int blocks = ((M_ / 32) * (E_ / 32)) / 8;
    gemm_kernel<<<blocks, 256, 0, stream>>>(normed, Wq_b, bq, nullptr,
                                            nullptr, Qb, M_, E_, E_, 1, 0);
    gemm_kernel<<<blocks, 256, 0, stream>>>(normed, Wk_b, bk, nullptr,
                                            nullptr, Kbuf, M_, E_, E_, 1, 0);
    gemm_kernel<<<blocks, 256, 0, stream>>>(normed, Wv_b, bv, nullptr,
                                            nullptr, Vt, M_, E_, E_, 2, 0);
  }

  // 4) fused attention (scores + softmax in LDS) + residual -> x2
  attn_kernel<<<M_ / 16, 256, 0, stream>>>(Qb, Kbuf, Vt, mask, x, x2);

  // 5) LN2 (reuse normed buffer)
  ln_kernel<<<M_, 256, 0, stream>>>(x2, ln2w, ln2b, normed);

  // 6) FFN1: relu(normed * W1^T + b1) -> h (bf16)
  gemm_kernel<<<((M_ / 32) * (F_ / 32)) / 8, 256, 0, stream>>>(
      normed, W1_b, b1, nullptr, nullptr, h, M_, F_, E_, 1, 1);

  // 7) FFN2: x2 + h * W2^T + b2 -> out (fp32)
  gemm_kernel<<<((M_ / 32) * (E_ / 32)) / 8, 256, 0, stream>>>(
      h, W2_b, b2, x2, out, nullptr, M_, E_, F_, 0, 0);
}
